// RNNEstimator_32890859553556
// MI455X (gfx1250) — matrix operand order repaired and checked by reading, not compiled, not run on hardware
//
#include <hip/hip_runtime.h>
#include <hip/hip_bf16.h>
#include <math.h>

// ---------------------------------------------------------------------------
// CDNA5 (gfx1250) persistent fused GRU estimator, v4.
//   - wave32, WMMA 16x16x32 bf16
//   - batch tiled 16/WG (32 WGs), recurrence fully local per WG
//   - A fragments loaded inline from LDS (no register arrays -> no scratch)
//   - belief kept as bf16 ping-pong in LDS; GRU update in registers
//   - native f32<->bf16 conversions; obs prefetch for the next step
// ---------------------------------------------------------------------------

typedef __attribute__((ext_vector_type(16))) __bf16 v16bf;
typedef __attribute__((ext_vector_type(8)))  __bf16 v8bf;
typedef __attribute__((ext_vector_type(8)))  float  v8f;

#define T_STEPS 1000
#define BATCH   512
#define TILE    16
#define HID     256
#define G3      768
#define MIN_STD 1e-4f

__device__ __forceinline__ __bf16 f2bf(float f) { return (__bf16)f; }   // native cvt
__device__ __forceinline__ float  bf2f(__bf16 b) { return (float)b; }

__device__ __forceinline__ v8f wmma_bf16(v16bf a, v16bf b, v8f c) {
    return __builtin_amdgcn_wmma_f32_16x16x32_bf16(
        false, a, false, b, (short)0, c, false, false);
}

// A fragment (contiguous k order, same as the B loads) from a bf16 LDS row of the A tile.
// lane row m = lane&15; kbase = (lane&16)?16:0; elems [kb..kb+15] contiguous.
__device__ __forceinline__ v16bf load_a_frag(const __bf16* row, int kt, int lane) {
    const int kb = (lane & 16) ? 16 : 0;
    const __bf16* p = row + 32 * kt + kb;
    union { v16bf v; v8bf h[2]; } u;
    u.h[0] = *(const v8bf*)(p);
    u.h[1] = *(const v8bf*)(p + 8);
    return u.v;
}

struct Smem {
    __bf16 belief[2][TILE][HID]; // ping-pong bf16 hidden state  16 KB
    __bf16 embed[TILE][HID];     // A tile for gi / decoder d      8 KB
    __bf16 h1[TILE][HID];        // enc layer-1 activations        8 KB
    float  x[TILE][4];           // delta features
    float  w1[HID][4];           // enc_w1 cached                  4 KB
    float  b1[HID], b2[HID];
    float  bih[G3], bhh[G3];
    float  db1[HID], db2[16];
};                               // ~45.3 KB

__global__ void cvt_bf16_kernel(const float* __restrict__ src,
                                __bf16* __restrict__ dst, int n) {
    int i = blockIdx.x * 256 + threadIdx.x;
    if (i < n) dst[i] = f2bf(src[i]);
}

__global__ __launch_bounds__(256)
void gru_persistent_kernel(const float* __restrict__ obs,
                           const float* __restrict__ enc_w1,
                           const float* __restrict__ enc_b1,
                           const float* __restrict__ enc_b2,
                           const float* __restrict__ gru_b_ih,
                           const float* __restrict__ gru_b_hh,
                           const float* __restrict__ dec_b1,
                           const float* __restrict__ dec_b2,
                           const __bf16* __restrict__ w_enc2,   // [256][256] bf16
                           const __bf16* __restrict__ w_ih,     // [768][256] bf16
                           const __bf16* __restrict__ w_hh,     // [768][256] bf16
                           const __bf16* __restrict__ w_dec1,   // [256][256] bf16
                           const __bf16* __restrict__ w_dec2,   // [16][256]  bf16
                           float* __restrict__ out) {
    __shared__ Smem s;
    const int tid  = threadIdx.x;
    const int lane = tid & 31;
    const int wave = tid >> 5;
    const int m0   = blockIdx.x * TILE;           // batch tile base

    // ---- cache small fp32 params in LDS, zero belief[0] ----
    for (int i = tid; i < HID * 4; i += 256) ((float*)s.w1)[i] = enc_w1[i];
    for (int i = tid; i < HID; i += 256) {
        s.b1[i] = enc_b1[i]; s.b2[i] = enc_b2[i]; s.db1[i] = dec_b1[i];
    }
    for (int i = tid; i < G3; i += 256) { s.bih[i] = gru_b_ih[i]; s.bhh[i] = gru_b_hh[i]; }
    if (tid < 16) s.db2[tid] = dec_b2[tid];
    {
        const __bf16 z0 = f2bf(0.f);
        for (int i = tid; i < TILE * HID; i += 256) ((__bf16*)s.belief[0])[i] = z0;
    }
    __syncthreads();

    const int mA     = lane & 15;                 // A row / C column index
    const int rowOff = (lane & 16) ? 8 : 0;       // C row offset
    const int klo    = (lane & 16) ? 16 : 0;      // B fragment K base (elements)
    const size_t g1  = (size_t)256 * HID, g2 = (size_t)512 * HID;

    // channel map for delta features: obs idx {3,4,6,7}
    const int chmap[4] = {3, 4, 6, 7};

    for (int t = 0; t < T_STEPS; ++t) {
        const int rd = t & 1, wr = rd ^ 1;

        // ---- delta features: 64 threads, one diff each ----
        if (tid < TILE * 4) {
            const int m = tid >> 2, c = tid & 3;
            const float* o0 = obs + ((size_t)t * BATCH + (m0 + m)) * 10 + chmap[c];
            const float* o1 = o0 + (size_t)BATCH * 10;
            s.x[m][c] = *o0 - *o1;
            // prefetch next step's rows into cache (speculative)
            __builtin_prefetch(o1 + (size_t)BATCH * 10, 0, 1);
        }
        __syncthreads();

        // ---- enc layer 1 (K=4): pure VALU ----
        {
            const int m = tid >> 4, g = tid & 15;
            const float x0 = s.x[m][0], x1 = s.x[m][1], x2 = s.x[m][2], x3 = s.x[m][3];
#pragma unroll
            for (int i = 0; i < 16; ++i) {
                const int j = g * 16 + i;
                float a = s.b1[j] + x0 * s.w1[j][0] + x1 * s.w1[j][1]
                                  + x2 * s.w1[j][2] + x3 * s.w1[j][3];
                s.h1[m][j] = f2bf(fmaxf(a, 0.f));
            }
        }
        __syncthreads();

        // ---- enc layer 2 (WMMA): embed[16x256]; wave owns 2 N-tiles ----
        {
            const int col0 = wave * 32 + mA;
            const int col1 = col0 + 16;
            const __bf16* wp0 = w_enc2 + (size_t)col0 * HID + klo;
            const __bf16* wp1 = w_enc2 + (size_t)col1 * HID + klo;
            v8f acc0, acc1;
            {
                const float b0 = s.b2[col0], b1v = s.b2[col1];
#pragma unroll
                for (int r = 0; r < 8; ++r) { acc0[r] = b0; acc1[r] = b1v; }
            }
#pragma unroll 1
            for (int kt = 0; kt < 8; ++kt) {
                const v16bf a = load_a_frag(&s.h1[mA][0], kt, lane);
                const int o = 32 * kt;
                acc0 = wmma_bf16(a, *(const v16bf*)(wp0 + o), acc0);
                acc1 = wmma_bf16(a, *(const v16bf*)(wp1 + o), acc1);
            }
#pragma unroll
            for (int r = 0; r < 8; ++r) {
                s.embed[r + rowOff][col0] = f2bf(acc0[r]);
                s.embed[r + rowOff][col1] = f2bf(acc1[r]);
            }
        }
        __syncthreads();

        // ---- fused GRU gates + update; wave owns hidden slice [32w, 32w+32) ----
        {
            const int jc0 = wave * 32 + mA;
            const int jc1 = jc0 + 16;
            const __bf16* wi0 = w_ih + (size_t)jc0 * HID + klo;
            const __bf16* wi1 = w_ih + (size_t)jc1 * HID + klo;
            const __bf16* wh0 = w_hh + (size_t)jc0 * HID + klo;
            const __bf16* wh1 = w_hh + (size_t)jc1 * HID + klo;

            v8f aR0, aZ0, aN0, aR1, aZ1, aN1, aH0, aH1;
            {
                const float bR0 = s.bih[jc0]       + s.bhh[jc0];
                const float bZ0 = s.bih[jc0 + 256] + s.bhh[jc0 + 256];
                const float bN0 = s.bih[jc0 + 512];
                const float bR1 = s.bih[jc1]       + s.bhh[jc1];
                const float bZ1 = s.bih[jc1 + 256] + s.bhh[jc1 + 256];
                const float bN1 = s.bih[jc1 + 512];
                const float bH0 = s.bhh[jc0 + 512];
                const float bH1 = s.bhh[jc1 + 512];
#pragma unroll
                for (int r = 0; r < 8; ++r) {
                    aR0[r] = bR0; aZ0[r] = bZ0; aN0[r] = bN0;
                    aR1[r] = bR1; aZ1[r] = bZ1; aN1[r] = bN1;
                    aH0[r] = bH0; aH1[r] = bH1;
                }
            }

            // e-pass: contributions from embed (i_r, i_z, i_n)
#pragma unroll 1
            for (int kt = 0; kt < 8; ++kt) {
                const v16bf e = load_a_frag(&s.embed[mA][0], kt, lane);
                const int o = 32 * kt;
                aR0 = wmma_bf16(e, *(const v16bf*)(wi0 + o),      aR0);
                aZ0 = wmma_bf16(e, *(const v16bf*)(wi0 + g1 + o), aZ0);
                aN0 = wmma_bf16(e, *(const v16bf*)(wi0 + g2 + o), aN0);
                aR1 = wmma_bf16(e, *(const v16bf*)(wi1 + o),      aR1);
                aZ1 = wmma_bf16(e, *(const v16bf*)(wi1 + g1 + o), aZ1);
                aN1 = wmma_bf16(e, *(const v16bf*)(wi1 + g2 + o), aN1);
            }
            __builtin_amdgcn_sched_barrier(0);

            // h-pass: contributions from belief[rd] (h_r, h_z, h_n)
#pragma unroll 1
            for (int kt = 0; kt < 8; ++kt) {
                const v16bf h = load_a_frag(&s.belief[rd][mA][0], kt, lane);
                const int o = 32 * kt;
                aR0 = wmma_bf16(h, *(const v16bf*)(wh0 + o),      aR0);
                aZ0 = wmma_bf16(h, *(const v16bf*)(wh0 + g1 + o), aZ0);
                aH0 = wmma_bf16(h, *(const v16bf*)(wh0 + g2 + o), aH0);
                aR1 = wmma_bf16(h, *(const v16bf*)(wh1 + o),      aR1);
                aZ1 = wmma_bf16(h, *(const v16bf*)(wh1 + g1 + o), aZ1);
                aH1 = wmma_bf16(h, *(const v16bf*)(wh1 + g2 + o), aH1);
            }
            __builtin_amdgcn_sched_barrier(0);

            // in-register GRU nonlinearity + state update (read rd, write wr)
#pragma unroll
            for (int r = 0; r < 8; ++r) {
                const int m = r + rowOff;
                {
                    const float rg = 1.f / (1.f + __expf(-aR0[r]));
                    const float zg = 1.f / (1.f + __expf(-aZ0[r]));
                    const float ng = tanhf(aN0[r] + rg * aH0[r]);
                    const float b  = bf2f(s.belief[rd][m][jc0]);
                    s.belief[wr][m][jc0] = f2bf((1.f - zg) * ng + zg * b);
                }
                {
                    const float rg = 1.f / (1.f + __expf(-aR1[r]));
                    const float zg = 1.f / (1.f + __expf(-aZ1[r]));
                    const float ng = tanhf(aN1[r] + rg * aH1[r]);
                    const float b  = bf2f(s.belief[rd][m][jc1]);
                    s.belief[wr][m][jc1] = f2bf((1.f - zg) * ng + zg * b);
                }
            }
        }
        // no trailing barrier: next step's barriers fence belief[wr] before reuse
    }
    __syncthreads();

    // ---- decoder layer 1 (WMMA, relu) -> reuse s.embed as "d" tile ----
    // final belief is in buffer T_STEPS & 1 == 0
    {
        const int col0 = wave * 32 + mA;
        const int col1 = col0 + 16;
        const __bf16* wp0 = w_dec1 + (size_t)col0 * HID + klo;
        const __bf16* wp1 = w_dec1 + (size_t)col1 * HID + klo;
        v8f acc0, acc1;
        {
            const float b0 = s.db1[col0], b1v = s.db1[col1];
#pragma unroll
            for (int r = 0; r < 8; ++r) { acc0[r] = b0; acc1[r] = b1v; }
        }
#pragma unroll 1
        for (int kt = 0; kt < 8; ++kt) {
            const v16bf a = load_a_frag(&s.belief[0][mA][0], kt, lane);
            const int o = 32 * kt;
            acc0 = wmma_bf16(a, *(const v16bf*)(wp0 + o), acc0);
            acc1 = wmma_bf16(a, *(const v16bf*)(wp1 + o), acc1);
        }
#pragma unroll
        for (int r = 0; r < 8; ++r) {
            s.embed[r + rowOff][col0] = f2bf(fmaxf(acc0[r], 0.f));
            s.embed[r + rowOff][col1] = f2bf(fmaxf(acc1[r], 0.f));
        }
    }
    __syncthreads();

    // ---- decoder layer 2 (16 outputs): wave 0 only; write means/stds ----
    if (wave == 0) {
        const int col = mA;
        const __bf16* wp = w_dec2 + (size_t)col * HID + klo;
        v8f acc;
        {
            const float bias = s.db2[col];
#pragma unroll
            for (int r = 0; r < 8; ++r) acc[r] = bias;
        }
#pragma unroll 1
        for (int kt = 0; kt < 8; ++kt) {
            const v16bf a = load_a_frag(&s.embed[mA][0], kt, lane);
            acc = wmma_bf16(a, *(const v16bf*)(wp + 32 * kt), acc);
        }
#pragma unroll
        for (int r = 0; r < 8; ++r) {
            const int bg = m0 + r + rowOff;                  // global batch row
            const float v = acc[r];
            if (col < 8) {
                out[(size_t)bg * 8 + col] = v;               // means
            } else {
                const float sp = (v > 20.f) ? v : log1pf(__expf(v));
                out[(size_t)BATCH * 8 + (size_t)bg * 8 + (col - 8)] = MIN_STD + sp;
            }
        }
    }
}

// ---------------------------------------------------------------------------
extern "C" void kernel_launch(void* const* d_in, const int* in_sizes, int n_in,
                              void* d_out, int out_size, void* d_ws, size_t ws_size,
                              hipStream_t stream) {
    const float* obs      = (const float*)d_in[0];
    // d_in[1] = act (unused by the reference computation)
    const float* enc_w1   = (const float*)d_in[2];
    const float* enc_b1   = (const float*)d_in[3];
    const float* enc_w2   = (const float*)d_in[4];
    const float* enc_b2   = (const float*)d_in[5];
    const float* gru_w_ih = (const float*)d_in[6];
    const float* gru_b_ih = (const float*)d_in[7];
    const float* gru_w_hh = (const float*)d_in[8];
    const float* gru_b_hh = (const float*)d_in[9];
    const float* dec_w1   = (const float*)d_in[10];
    const float* dec_b1   = (const float*)d_in[11];
    const float* dec_w2   = (const float*)d_in[12];
    const float* dec_b2   = (const float*)d_in[13];
    float* out = (float*)d_out;

    // bf16 weight arena in workspace: [out][in] row-major, WMMA-B friendly
    __bf16* wbf = (__bf16*)d_ws;
    __bf16* w_enc2 = wbf;                 // 256*256
    __bf16* w_ih   = wbf + 65536;         // 768*256
    __bf16* w_hh   = wbf + 65536 + 196608;
    __bf16* w_dec1 = wbf + 65536 + 2 * 196608;
    __bf16* w_dec2 = wbf + 2 * 65536 + 2 * 196608;

    auto cvt = [&](const float* src, __bf16* dst, int n) {
        cvt_bf16_kernel<<<(n + 255) / 256, 256, 0, stream>>>(src, dst, n);
    };
    cvt(enc_w2,   w_enc2, 256 * 256);
    cvt(gru_w_ih, w_ih,   768 * 256);
    cvt(gru_w_hh, w_hh,   768 * 256);
    cvt(dec_w1,   w_dec1, 256 * 256);
    cvt(dec_w2,   w_dec2, 16 * 256);

    gru_persistent_kernel<<<dim3(BATCH / TILE), dim3(256), 0, stream>>>(
        obs, enc_w1, enc_b1, enc_b2, gru_b_ih, gru_b_hh, dec_b1, dec_b2,
        w_enc2, w_ih, w_hh, w_dec1, w_dec2, out);
}
